// MSIGNN_50010599194662
// MI455X (gfx1250) — compile-verified, matrix operand order checked by reading, not run on hardware
//
#include <hip/hip_runtime.h>
#include <hip/hip_bf16.h>

// ---------------------------------------------------------------------------
// MI455X (gfx1250) fused GAT + edge-MLP pipeline.
// Heavy math runs through v_wmma_f32_16x16x32_f16 (f16 in, f32 accum).
// Edge MLP (dominant ~336 GFLOP) is fully fused: feat/hidden tiles live in
// LDS as f16 (96 KB dynamic LDS, WGP has 320 KB); each B fragment is reused
// across 2 M-tiles in registers to halve L2 weight traffic.
// ---------------------------------------------------------------------------

typedef __attribute__((ext_vector_type(16))) _Float16 v16h;
typedef __attribute__((ext_vector_type(8)))  _Float16 v8h;
typedef __attribute__((ext_vector_type(8)))  float    v8f;

// Problem constants (from the reference)
#define F0      128
#define F1      128
#define H1c     2
#define F2c     128
#define H2c     4
#define FIN2    512     // H1*(F0+F1)
#define FFN_IN  512     // 4*F2
#define FFN_H1  1024
#define FFN_H2  512
#define TILE_E  32      // edges per block in fused MLP

__device__ __forceinline__ v8f wmma16(v16h a, v16h b, v8f c) {
    return __builtin_amdgcn_wmma_f32_16x16x32_f16(
        /*neg_a=*/false, a, /*neg_b=*/false, b,
        /*c_mod=*/(short)0, c, /*reuse_a=*/false, /*reuse_b=*/false);
}

// Load one 16-lane fragment (A or B operand of 16x16x32 f16 WMMA) given the
// per-lane base pointer: halves K..K+7 at p, K+16..K+23 at p+16 (ISA layout).
__device__ __forceinline__ v16h ld_frag(const _Float16* p) {
    v8h lo = *(const v8h*)(p);
    v8h hi = *(const v8h*)(p + 16);
    v16h r;
#pragma unroll
    for (int i = 0; i < 8; ++i) { r[i] = lo[i]; r[i + 8] = hi[i]; }
    return r;
}

// ---------------------------------------------------------------------------
// Elementwise helpers: f32->f16 copy, f32 [K,N] -> f16 [N,K] transpose
// ---------------------------------------------------------------------------
__global__ void k_f32_to_f16(const float* __restrict__ a, _Float16* __restrict__ b, int n) {
    int i = blockIdx.x * blockDim.x + threadIdx.x;
    if (i < n) b[i] = (_Float16)a[i];
}

__global__ void k_transpose_f16(const float* __restrict__ W, _Float16* __restrict__ Wt,
                                int K, int N) {
    int i = blockIdx.x * blockDim.x + threadIdx.x;
    if (i >= K * N) return;
    int k = i / N, n = i % N;
    Wt[(size_t)n * K + k] = (_Float16)W[i];
}

// ---------------------------------------------------------------------------
// Generic WMMA GEMM: C[M,N] = A[M,K](f16,row) @ Bt[N,K](f16,N-major)
// Requires M%16==0, N%64==0, K%32==0.  One wave -> 16x64 C strip.
// ---------------------------------------------------------------------------
__global__ __launch_bounds__(256)
void k_wmma_gemm(const _Float16* __restrict__ A, const _Float16* __restrict__ Bt,
                 float* __restrict__ C, int M, int N, int K) {
    const int lane = threadIdx.x & 31;
    const int wid  = (blockIdx.x * (blockDim.x >> 5)) + (threadIdx.x >> 5);
    const int Mt   = M >> 4;
    const int Ng   = N >> 6;          // groups of 4 n-tiles
    if (wid >= Mt * Ng) return;       // wave-uniform: EXEC stays full
    const int mt = wid / Ng, ng = wid % Ng;
    const int m0 = mt << 4;
    const int half = lane >> 4, l = lane & 15;

    v8f acc[4] = {};
    const _Float16* arow = A + (size_t)(m0 + l) * K + half * 8;
    const _Float16* brow = Bt + (size_t)((ng << 6) + l) * K + half * 8;

    for (int k0 = 0; k0 < K; k0 += 32) {
        v16h af = ld_frag(arow + k0);
#pragma unroll
        for (int j = 0; j < 4; ++j) {
            v16h bf = ld_frag(brow + (size_t)(j << 4) * K + k0);
            acc[j] = wmma16(af, bf, acc[j]);
        }
    }
#pragma unroll
    for (int j = 0; j < 4; ++j) {
        const int n0 = (ng << 6) + (j << 4);
        float* crow = C + (size_t)(m0 + half * 8) * N + n0 + l;
#pragma unroll
        for (int r = 0; r < 8; ++r) crow[(size_t)r * N] = acc[j][r];
    }
}

// ---------------------------------------------------------------------------
// GAT attention pieces
// ---------------------------------------------------------------------------
__global__ void k_scores(const float* __restrict__ proj, const float* __restrict__ a_src,
                         const float* __restrict__ a_trg, float* __restrict__ ssrc,
                         float* __restrict__ strg, int N, int H, int F) {
    int i = blockIdx.x * blockDim.x + threadIdx.x;
    if (i >= N * H) return;
    int n = i / H, h = i % H;
    const float* p = proj + (size_t)n * H * F + (size_t)h * F;
    float s1 = 0.f, s2 = 0.f;
    for (int f = 0; f < F; ++f) {
        float v = p[f];
        s1 += v * a_src[h * F + f];
        s2 += v * a_trg[h * F + f];
    }
    ssrc[i] = s1; strg[i] = s2;
}

__global__ void k_edge_att(const int* __restrict__ src, const int* __restrict__ trg,
                           const float* __restrict__ ssrc, const float* __restrict__ strg,
                           float* __restrict__ ex, float* __restrict__ denom, int E, int H) {
    int i = blockIdx.x * blockDim.x + threadIdx.x;
    if (i >= E * H) return;
    int e = i / H, h = i % H;
    float v = ssrc[src[e] * H + h] + strg[trg[e] * H + h];
    v = (v > 0.f) ? v : 0.2f * v;                // leaky relu
    float x = __expf(v);                          // global-max shift is a softmax no-op
    ex[i] = x;
    atomicAdd(&denom[trg[e] * H + h], x);
}

__global__ void k_edge_scatter(const int* __restrict__ src, const int* __restrict__ trg,
                               const float* __restrict__ proj, const float* __restrict__ ex,
                               const float* __restrict__ denom, float* __restrict__ agg,
                               int E, int H, int F) {
    int i = blockIdx.x * blockDim.x + threadIdx.x;
    if (i >= E * H * F) return;
    int f  = i % F;
    int eh = i / F;
    int h  = eh % H;
    int e  = eh / H;
    int t  = trg[e];
    float att = ex[eh] / (denom[t * H + h] + 1e-16f);
    float m = proj[(size_t)src[e] * H * F + (size_t)h * F + f] * att;
    atomicAdd(&agg[(size_t)t * H * F + (size_t)h * F + f], m);
}

// Layer1 epilogue: concat(agg, local_emb) per head, +b1, ELU, -> f16 [N,512]
__global__ void k_finish1(const float* __restrict__ agg, const float* __restrict__ lemb,
                          const float* __restrict__ b1, _Float16* __restrict__ h_h, int N) {
    int i = blockIdx.x * blockDim.x + threadIdx.x;
    if (i >= N * FIN2) return;
    int n = i / FIN2, c = i % FIN2;
    int hh = c / (F0 + F1), cc = c % (F0 + F1);
    float v = (cc < F1) ? agg[(size_t)n * (H1c * F1) + hh * F1 + cc]
                        : lemb[(size_t)n * F0 + (cc - F1)];
    v += b1[c];
    v = (v > 0.f) ? v : (__expf(v) - 1.f);       // ELU
    h_h[i] = (_Float16)v;
}

// Layer2 epilogue: mean over 4 heads, +b2, ELU -> f32 [N,128]
__global__ void k_finish2(const float* __restrict__ agg2, const float* __restrict__ b2,
                          float* __restrict__ outn, int N) {
    int i = blockIdx.x * blockDim.x + threadIdx.x;
    if (i >= N * F2c) return;
    int n = i / F2c, f = i % F2c;
    float s = 0.f;
#pragma unroll
    for (int h = 0; h < H2c; ++h) s += agg2[(size_t)n * (H2c * F2c) + h * F2c + f];
    s = s * 0.25f + b2[f];
    s = (s > 0.f) ? s : (__expf(s) - 1.f);
    outn[i] = s;
}

// ---------------------------------------------------------------------------
// Fused edge MLP: per block = 32 edges.
//   feat[32,512] (LDS f16) -> h1[32,1024] -> h2[32,512] -> sigmoid logit
// Dynamic LDS: sH1 (64 KB) + sF (32 KB, feat then reused as h2) = 96 KB.
// Each wave owns 2 M-tiles so every B fragment load feeds 2 WMMAs.
// ---------------------------------------------------------------------------
__global__ __launch_bounds__(256)
void k_edge_mlp(const float* __restrict__ outn, const int* __restrict__ src,
                const int* __restrict__ trg,
                const _Float16* __restrict__ Wi_t,  const float* __restrict__ bi,
                const _Float16* __restrict__ Wf1_t, const float* __restrict__ bf1,
                const float* __restrict__ Wf2,      const float* __restrict__ bf2,
                float* __restrict__ result, int E) {
    extern __shared__ _Float16 smem[];
    _Float16* sH1 = smem;                         // [32][1024]  64 KB
    _Float16* sF  = smem + TILE_E * FFN_H1;       // [32][512]   32 KB (feat -> h2)

    const int tid  = threadIdx.x;
    const int lane = tid & 31;
    const int w    = tid >> 5;
    const int e0   = blockIdx.x * TILE_E;

    // --- Phase A: build feature tile [32, 512] = [e1+e2 | e1*e2 | e1 | e2]
    {
        const int row = tid >> 3, c0 = tid & 7;   // 32 rows x 8 threads
        const int e = e0 + row;
        const int s = src[e], t = trg[e];
        const float* p1 = outn + (size_t)s * F2c;
        const float* p2 = outn + (size_t)t * F2c;
        _Float16* fr = sF + row * FFN_IN;
#pragma unroll
        for (int c = c0; c < F2c; c += 8) {
            float e1 = p1[c], e2 = p2[c];
            fr[c]             = (_Float16)(e1 + e2);
            fr[F2c + c]       = (_Float16)(e1 * e2);
            fr[2 * F2c + c]   = (_Float16)e1;
            fr[3 * F2c + c]   = (_Float16)e2;
        }
    }
    __syncthreads();

    const int half = lane >> 4, l = lane & 15;

    // --- Phase B: h1 = relu(feat @ Wi + bi)  [32,1024]
    //     wave w owns n-tiles w*8..w*8+7, and both M-tiles (rows 0-15, 16-31)
    {
        v8f acc[2][8];
#pragma unroll
        for (int j = 0; j < 8; ++j) {
            float b = bi[(w * 8 + j) * 16 + l];
#pragma unroll
            for (int r = 0; r < 8; ++r) { acc[0][j][r] = b; acc[1][j][r] = b; }
        }
        const _Float16* arow0 = sF + l * FFN_IN + half * 8;
        const _Float16* arow1 = sF + (16 + l) * FFN_IN + half * 8;
        const _Float16* brow  = Wi_t + (size_t)(w * 128 + l) * FFN_IN + half * 8;
        for (int k0 = 0; k0 < FFN_IN; k0 += 32) {
            v16h a0 = ld_frag(arow0 + k0);
            v16h a1 = ld_frag(arow1 + k0);
            if (k0 + 32 < FFN_IN)                 // hint next weight panel
                __builtin_prefetch(brow + k0 + 32, 0, 3);
#pragma unroll
            for (int j = 0; j < 8; ++j) {
                v16h bf = ld_frag(brow + (size_t)(j * 16) * FFN_IN + k0);
                acc[0][j] = wmma16(a0, bf, acc[0][j]);   // B frag reused x2
                acc[1][j] = wmma16(a1, bf, acc[1][j]);
            }
        }
#pragma unroll
        for (int m = 0; m < 2; ++m) {
#pragma unroll
            for (int j = 0; j < 8; ++j) {
                const int n0 = (w * 8 + j) * 16;
#pragma unroll
                for (int r = 0; r < 8; ++r) {
                    float v = acc[m][j][r];
                    v = (v > 0.f) ? v : 0.f;
                    sH1[(m * 16 + half * 8 + r) * FFN_H1 + n0 + l] = (_Float16)v;
                }
            }
        }
    }
    __syncthreads();   // sH1 complete; all reads of sF (feat) done

    // --- Phase C: h2 = relu(h1 @ Wf1 + bf1)  [32,512]
    //     wave w owns n-tiles w*4..w*4+3, both M-tiles; writes h2 into sF
    {
        v8f acc[2][4];
#pragma unroll
        for (int j = 0; j < 4; ++j) {
            float b = bf1[(w * 4 + j) * 16 + l];
#pragma unroll
            for (int r = 0; r < 8; ++r) { acc[0][j][r] = b; acc[1][j][r] = b; }
        }
        const _Float16* arow0 = sH1 + l * FFN_H1 + half * 8;
        const _Float16* arow1 = sH1 + (16 + l) * FFN_H1 + half * 8;
        const _Float16* brow  = Wf1_t + (size_t)(w * 64 + l) * FFN_H1 + half * 8;
        for (int k0 = 0; k0 < FFN_H1; k0 += 32) {
            v16h a0 = ld_frag(arow0 + k0);
            v16h a1 = ld_frag(arow1 + k0);
            if (k0 + 32 < FFN_H1)
                __builtin_prefetch(brow + k0 + 32, 0, 3);
#pragma unroll
            for (int j = 0; j < 4; ++j) {
                v16h bf = ld_frag(brow + (size_t)(j * 16) * FFN_H1 + k0);
                acc[0][j] = wmma16(a0, bf, acc[0][j]);
                acc[1][j] = wmma16(a1, bf, acc[1][j]);
            }
        }
#pragma unroll
        for (int m = 0; m < 2; ++m) {
#pragma unroll
            for (int j = 0; j < 4; ++j) {
                const int n0 = (w * 4 + j) * 16;
#pragma unroll
                for (int r = 0; r < 8; ++r) {
                    float v = acc[m][j][r];
                    v = (v > 0.f) ? v : 0.f;
                    sF[(m * 16 + half * 8 + r) * FFN_H2 + n0 + l] = (_Float16)v;
                }
            }
        }
    }
    __syncthreads();

    // --- Phase D: logit = h2 . Wf2 + bf2; sigmoid.  8 threads reduce each row.
    {
        const int row = tid >> 3, c0 = tid & 7;
        float s = 0.f;
        const _Float16* hr = sF + row * FFN_H2;
#pragma unroll 8
        for (int c = c0; c < FFN_H2; c += 8) s += (float)hr[c] * Wf2[c];
#pragma unroll
        for (int m = 4; m >= 1; m >>= 1) s += __shfl_xor(s, m, 32);  // within 8-lane group
        if (c0 == 0) {
            float logit = s + bf2[0];
            result[e0 + row] = 1.f / (1.f + __expf(-logit));
        }
    }
}

// ---------------------------------------------------------------------------
// Host launch
// ---------------------------------------------------------------------------
extern "C" void kernel_launch(void* const* d_in, const int* in_sizes, int n_in,
                              void* d_out, int out_size, void* d_ws, size_t ws_size,
                              hipStream_t stream) {
    const float* x     = (const float*)d_in[0];
    const int*   eidx  = (const int*)  d_in[1];
    const float* lemb  = (const float*)d_in[2];
    const float* W1    = (const float*)d_in[3];
    const float* asrc1 = (const float*)d_in[4];
    const float* atrg1 = (const float*)d_in[5];
    const float* b1    = (const float*)d_in[6];
    const float* W2    = (const float*)d_in[7];
    const float* asrc2 = (const float*)d_in[8];
    const float* atrg2 = (const float*)d_in[9];
    const float* b2    = (const float*)d_in[10];
    const float* Wi    = (const float*)d_in[11];
    const float* bi    = (const float*)d_in[12];
    const float* Wf1   = (const float*)d_in[13];
    const float* bf1   = (const float*)d_in[14];
    const float* Wf2   = (const float*)d_in[15];
    const float* bf2   = (const float*)d_in[16];

    const int N = in_sizes[0] / F0;       // 10000
    const int E = in_sizes[1] / 2;        // 160000
    const int* src = eidx;
    const int* trg = eidx + E;
    float* result = (float*)d_out;

    // --- workspace carve-out (256B aligned)
    char* ws = (char*)d_ws;
    size_t off = 0;
    auto alloc = [&](size_t bytes) -> char* {
        char* p = ws + off;
        off = (off + bytes + 255) & ~(size_t)255;
        return p;
    };
    _Float16* x_h    = (_Float16*)alloc((size_t)N * F0 * 2);
    _Float16* W1_t   = (_Float16*)alloc((size_t)(H1c * F1) * F0 * 2);
    float*    proj1  = (float*)   alloc((size_t)N * H1c * F1 * 4);
    float*    ssrc1  = (float*)   alloc((size_t)N * H1c * 4);
    float*    strg1  = (float*)   alloc((size_t)N * H1c * 4);
    float*    ex1    = (float*)   alloc((size_t)E * H1c * 4);
    float*    den1   = (float*)   alloc((size_t)N * H1c * 4);
    float*    agg1   = (float*)   alloc((size_t)N * H1c * F1 * 4);
    _Float16* h_h    = (_Float16*)alloc((size_t)N * FIN2 * 2);
    _Float16* W2_t   = (_Float16*)alloc((size_t)(H2c * F2c) * FIN2 * 2);
    float*    proj2  = (float*)   alloc((size_t)N * H2c * F2c * 4);
    float*    ssrc2  = (float*)   alloc((size_t)N * H2c * 4);
    float*    strg2  = (float*)   alloc((size_t)N * H2c * 4);
    float*    ex2    = (float*)   alloc((size_t)E * H2c * 4);
    float*    den2   = (float*)   alloc((size_t)N * H2c * 4);
    float*    agg2   = (float*)   alloc((size_t)N * H2c * F2c * 4);
    float*    outn   = (float*)   alloc((size_t)N * F2c * 4);
    _Float16* Wi_t   = (_Float16*)alloc((size_t)FFN_H1 * FFN_IN * 2);
    _Float16* Wf1_t  = (_Float16*)alloc((size_t)FFN_H2 * FFN_H1 * 2);
    if (off > ws_size) return;   // insufficient scratch

    const int TB = 256;
    auto blk = [&](long long n) { return (int)((n + TB - 1) / TB); };

    // --- weight / input conversion (every call; deterministic)
    k_f32_to_f16   <<<blk((long long)N * F0), TB, 0, stream>>>(x, x_h, N * F0);
    k_transpose_f16<<<blk((long long)F0 * H1c * F1), TB, 0, stream>>>(W1, W1_t, F0, H1c * F1);
    k_transpose_f16<<<blk((long long)FIN2 * H2c * F2c), TB, 0, stream>>>(W2, W2_t, FIN2, H2c * F2c);
    k_transpose_f16<<<blk((long long)FFN_IN * FFN_H1), TB, 0, stream>>>(Wi, Wi_t, FFN_IN, FFN_H1);
    k_transpose_f16<<<blk((long long)FFN_H1 * FFN_H2), TB, 0, stream>>>(Wf1, Wf1_t, FFN_H1, FFN_H2);

    // --- zero accumulation buffers (graph-capture safe)
    hipMemsetAsync(den1, 0, (size_t)N * H1c * 4, stream);
    hipMemsetAsync(agg1, 0, (size_t)N * H1c * F1 * 4, stream);
    hipMemsetAsync(den2, 0, (size_t)N * H2c * 4, stream);
    hipMemsetAsync(agg2, 0, (size_t)N * H2c * F2c * 4, stream);

    // --- GAT layer 1
    {
        const int M = N, Nc = H1c * F1, K = F0;           // 10000 x 256, K=128
        const int waves = (M >> 4) * (Nc >> 6);
        k_wmma_gemm<<<(waves + 7) / 8, 256, 0, stream>>>(x_h, W1_t, proj1, M, Nc, K);
    }
    k_scores      <<<blk((long long)N * H1c), TB, 0, stream>>>(proj1, asrc1, atrg1, ssrc1, strg1, N, H1c, F1);
    k_edge_att    <<<blk((long long)E * H1c), TB, 0, stream>>>(src, trg, ssrc1, strg1, ex1, den1, E, H1c);
    k_edge_scatter<<<blk((long long)E * H1c * F1), TB, 0, stream>>>(src, trg, proj1, ex1, den1, agg1, E, H1c, F1);
    k_finish1     <<<blk((long long)N * FIN2), TB, 0, stream>>>(agg1, lemb, b1, h_h, N);

    // --- GAT layer 2
    {
        const int M = N, Nc = H2c * F2c, K = FIN2;        // 10000 x 512, K=512
        const int waves = (M >> 4) * (Nc >> 6);
        k_wmma_gemm<<<(waves + 7) / 8, 256, 0, stream>>>(h_h, W2_t, proj2, M, Nc, K);
    }
    k_scores      <<<blk((long long)N * H2c), TB, 0, stream>>>(proj2, asrc2, atrg2, ssrc2, strg2, N, H2c, F2c);
    k_edge_att    <<<blk((long long)E * H2c), TB, 0, stream>>>(src, trg, ssrc2, strg2, ex2, den2, E, H2c);
    k_edge_scatter<<<blk((long long)E * H2c * F2c), TB, 0, stream>>>(src, trg, proj2, ex2, den2, agg2, E, H2c, F2c);
    k_finish2     <<<blk((long long)N * F2c), TB, 0, stream>>>(agg2, b2, outn, N);

    // --- fused edge MLP (dominant compute): 32 edges per block, 96 KB LDS
    const size_t lds_bytes = (size_t)(TILE_E * FFN_H1 + TILE_E * FFN_IN) * sizeof(_Float16);
    k_edge_mlp<<<E / TILE_E, 256, lds_bytes, stream>>>(outn, src, trg, Wi_t, bi,
                                                       Wf1_t, bf1, Wf2, bf2, result, E);
}
